// PartialBQQLinear_25589415149865
// MI455X (gfx1250) — matrix operand order repaired and checked by reading, not compiled, not run on hardware
//
#include <hip/hip_runtime.h>

// ---------------- problem constants (from reference) ----------------
constexpr int kBW = 4, kRW = 32, kCW = 32, kYR = 128, kID = 16, kZC = 128;
constexpr int kOUT = kRW * kYR;   // 4096
constexpr int kIN  = kCW * kZC;   // 4096
constexpr int kTOK = 8192;

typedef __bf16 v8bf  __attribute__((ext_vector_type(8)));
typedef __bf16 v16bf __attribute__((ext_vector_type(16)));
typedef float  v8f   __attribute__((ext_vector_type(8)));

__device__ __forceinline__ unsigned short f32_to_bf16_rne(float f) {
  unsigned int u = __float_as_uint(f);
  u += 0x7fffu + ((u >> 16) & 1u);   // round-to-nearest-even
  return (unsigned short)(u >> 16);
}
__device__ __forceinline__ float bf16_bits_to_f32(unsigned short h) {
  return __uint_as_float(((unsigned int)h) << 16);
}

// ---------------- kernel 1: split X (fp32) into bf16 hi/lo planes ----------------
__global__ __launch_bounds__(256) void convert_x_kernel(
    const float* __restrict__ X, __bf16* __restrict__ Xh, __bf16* __restrict__ Xl) {
  const size_t i = ((size_t)blockIdx.x * blockDim.x + threadIdx.x) * 4;
  const float4 v = *(const float4*)(X + i);
  float xs[4] = {v.x, v.y, v.z, v.w};
#pragma unroll
  for (int j = 0; j < 4; ++j) {
    unsigned short hb = f32_to_bf16_rne(xs[j]);
    unsigned short lb = f32_to_bf16_rne(xs[j] - bf16_bits_to_f32(hb));
    Xh[i + j] = __builtin_bit_cast(__bf16, hb);
    Xl[i + j] = __builtin_bit_cast(__bf16, lb);
  }
}

// ---------------- kernel 2: build W (BQQ or float_weight) as bf16 hi/lo ----------------
// one block per (rw, cw) 128x128 tile; blockIdx.x = cw, blockIdx.y = rw
__global__ __launch_bounds__(256) void build_w_kernel(
    const unsigned char* __restrict__ Y, const unsigned char* __restrict__ Z,
    const float* __restrict__ a, const float* __restrict__ b, const float* __restrict__ c,
    const float* __restrict__ d, const float* __restrict__ fw,
    const unsigned char* __restrict__ qmask,
    __bf16* __restrict__ Wh, __bf16* __restrict__ Wl) {
  const int cw = blockIdx.x, rw = blockIdx.y;
  const int tid = threadIdx.x;

  __shared__ unsigned char Ys[kBW][kYR][kID];   // 8 KB
  __shared__ unsigned char Zs[kBW][kID][kZC];   // 8 KB
  __shared__ float Ysum[kBW][kYR];              // 2 KB
  __shared__ float Zsum[kBW][kZC];              // 2 KB

  for (int e = tid; e < kBW * kYR * kID; e += 256) {
    int bw = e / (kYR * kID), rem = e % (kYR * kID);
    (&Ys[0][0][0])[e] = Y[((size_t)(bw * kRW + rw) * kCW + cw) * (kYR * kID) + rem];
  }
  for (int e = tid; e < kBW * kID * kZC; e += 256) {
    int bw = e / (kID * kZC), rem = e % (kID * kZC);
    (&Zs[0][0][0])[e] = Z[((size_t)(bw * kRW + rw) * kCW + cw) * (kID * kZC) + rem];
  }
  __syncthreads();
  for (int e = tid; e < kBW * kYR; e += 256) {
    int bw = e >> 7, yr = e & (kYR - 1);
    int s = 0;
    for (int id = 0; id < kID; ++id) s += Ys[bw][yr][id];
    Ysum[bw][yr] = (float)s;
  }
  for (int e = tid; e < kBW * kZC; e += 256) {
    int bw = e >> 7, zc = e & (kZC - 1);
    int s = 0;
    for (int id = 0; id < kID; ++id) s += Zs[bw][id][zc];
    Zsum[bw][zc] = (float)s;
  }
  __syncthreads();

  float av[kBW], bv[kBW], cv[kBW];
#pragma unroll
  for (int bw = 0; bw < kBW; ++bw) {
    size_t idx = (size_t)(bw * kRW + rw) * kCW + cw;
    av[bw] = a[idx]; bv[bw] = b[idx]; cv[bw] = c[idx];
  }
  const float dv = d[rw * kCW + cw];
  const bool quant = (qmask[rw * kCW + cw] != 0);

  const int zc = tid & (kZC - 1);
  const int yrbase = (tid >> 7) * 64;
  for (int yy = 0; yy < 64; ++yy) {
    const int yr = yrbase + yy;
    const size_t orow = (size_t)(rw * kYR + yr);   // out index
    const size_t ocol = (size_t)(cw * kZC + zc);   // in index
    float w;
    if (quant) {
      w = dv;
      for (int bw = 0; bw < kBW; ++bw) {
        int dot = 0;
#pragma unroll
        for (int id = 0; id < kID; ++id) dot += (int)Ys[bw][yr][id] * (int)Zs[bw][id][zc];
        w += av[bw] * (float)dot + bv[bw] * Ysum[bw][yr] + cv[bw] * Zsum[bw][zc];
      }
    } else {
      w = fw[orow * kIN + ocol];
    }
    unsigned short hb = f32_to_bf16_rne(w);
    unsigned short lb = f32_to_bf16_rne(w - bf16_bits_to_f32(hb));
    Wh[orow * kIN + ocol] = __builtin_bit_cast(__bf16, hb);
    Wl[orow * kIN + ocol] = __builtin_bit_cast(__bf16, lb);
  }
}

// ---------------- kernel 3: out = X @ W^T + bias via bf16x3 WMMA ----------------
// A fragment (16x32, 16-bit): lane holds row m0+(lane&15); elements 0..7 at
// K = k0 + (lane>=16 ? 8 : 0), elements 8..15 at +16 (per ISA VGPR layout).
__device__ __forceinline__ v16bf load_a_frag(const __bf16* p) {
  const v8bf x = *(const v8bf*)(p);
  const v8bf y = *(const v8bf*)(p + 16);
  return __builtin_shufflevector(x, y, 0, 1, 2, 3, 4, 5, 6, 7,
                                 8, 9, 10, 11, 12, 13, 14, 15);
}

__global__ __launch_bounds__(128) void bqq_gemm_kernel(
    const __bf16* __restrict__ Xh, const __bf16* __restrict__ Xl,
    const __bf16* __restrict__ Wh, const __bf16* __restrict__ Wl,
    const float* __restrict__ bias, float* __restrict__ out) {
  const int lane = threadIdx.x & 31;
  const int wave = threadIdx.x >> 5;
  // 4 waves in 2x2 -> block tile 128(M) x 128(N); wave tile 64(M) x 64(N)
  const int m0 = blockIdx.y * 128 + (wave >> 1) * 64;
  const int n0 = blockIdx.x * 128 + (wave & 1) * 64;

  // loop-invariant per-lane 32-bit element offsets (<< 2^31) so the backend can
  // use saddr-form global loads: uniform base (stepped by k0) + fixed voffset.
  unsigned aoff[4], boff[4];
#pragma unroll
  for (int i = 0; i < 4; ++i)
    aoff[i] = (unsigned)((m0 + i * 16 + (lane & 15)) * kIN + ((lane >> 4) << 3));
#pragma unroll
  for (int j = 0; j < 4; ++j)
    boff[j] = (unsigned)((n0 + j * 16 + (lane & 15)) * kIN + ((lane >> 4) << 3));

  v8f acc[4][4] = {};

  for (int k0 = 0; k0 < kIN; k0 += 32) {
    const __bf16* pxh = Xh + k0;   // uniform scalar part
    const __bf16* pxl = Xl + k0;
    const __bf16* pwh = Wh + k0;
    const __bf16* pwl = Wl + k0;

    v16bf ah[4], al[4];
#pragma unroll
    for (int i = 0; i < 4; ++i) {
      ah[i] = load_a_frag(pxh + aoff[i]);
      al[i] = load_a_frag(pxl + aoff[i]);
    }
    // process N in two halves: keeps B live range short; jj=1 loads overlap
    // the jj=0 WMMA burst in fresh registers (fewer WAR hazard NOPs)
#pragma unroll
    for (int jj = 0; jj < 2; ++jj) {
      v16bf bh[2], bl[2];
#pragma unroll
      for (int t = 0; t < 2; ++t) {
        bh[t] = load_a_frag(pwh + boff[jj * 2 + t]);
        bl[t] = load_a_frag(pwl + boff[jj * 2 + t]);
      }
#pragma unroll
      for (int i = 0; i < 4; ++i) {
#pragma unroll
        for (int t = 0; t < 2; ++t) {
          const int j = jj * 2 + t;
          acc[i][j] = __builtin_amdgcn_wmma_f32_16x16x32_bf16(
              false, ah[i], false, bh[t], (short)0, acc[i][j], false, false);
          acc[i][j] = __builtin_amdgcn_wmma_f32_16x16x32_bf16(
              false, ah[i], false, bl[t], (short)0, acc[i][j], false, false);
          acc[i][j] = __builtin_amdgcn_wmma_f32_16x16x32_bf16(
              false, al[i], false, bh[t], (short)0, acc[i][j], false, false);
        }
      }
    }
  }

  // C/D layout: VGPR r, lanes 0-15 -> M=r, lanes 16-31 -> M=r+8; N = lane&15
  const int moff = (lane >> 4) << 3;
#pragma unroll
  for (int i = 0; i < 4; ++i) {
#pragma unroll
    for (int j = 0; j < 4; ++j) {
      const int n = n0 + j * 16 + (lane & 15);
      const float bv = bias[n];
#pragma unroll
      for (int r = 0; r < 8; ++r) {
        const int m = m0 + i * 16 + moff + r;
        out[(size_t)m * kOUT + n] = acc[i][j][r] + bv;
      }
    }
  }
}

// ---------------- launch ----------------
extern "C" void kernel_launch(void* const* d_in, const int* in_sizes, int n_in,
                              void* d_out, int out_size, void* d_ws, size_t ws_size,
                              hipStream_t stream) {
  (void)in_sizes; (void)n_in; (void)out_size; (void)ws_size;
  const float*         X    = (const float*)d_in[0];
  const unsigned char* Yb   = (const unsigned char*)d_in[1];  // bool
  const unsigned char* Zb   = (const unsigned char*)d_in[2];  // bool
  const float*         a    = (const float*)d_in[3];
  const float*         b    = (const float*)d_in[4];
  const float*         c    = (const float*)d_in[5];
  const float*         d    = (const float*)d_in[6];
  const float*         fw   = (const float*)d_in[7];
  const float*         bias = (const float*)d_in[8];
  const unsigned char* qm   = (const unsigned char*)d_in[9];  // bool

  // workspace layout: Xh | Xl (8192x4096 bf16 each), Wh | Wl (4096x4096 bf16 each)
  __bf16* Xh = (__bf16*)d_ws;
  __bf16* Xl = Xh + (size_t)kTOK * kIN;
  __bf16* Wh = Xl + (size_t)kTOK * kIN;
  __bf16* Wl = Wh + (size_t)kOUT * kIN;

  convert_x_kernel<<<(kTOK * kIN) / (256 * 4), 256, 0, stream>>>(X, Xh, Xl);
  build_w_kernel<<<dim3(kCW, kRW), 256, 0, stream>>>(Yb, Zb, a, b, c, d, fw, qm, Wh, Wl);
  bqq_gemm_kernel<<<dim3(kOUT / 128, kTOK / 128), 128, 0, stream>>>(
      Xh, Xl, Wh, Wl, bias, (float*)d_out);
}
